// LSTMModel_505
// MI455X (gfx1250) — matrix operand order repaired and checked by reading, not compiled, not run on hardware
//
#include <hip/hip_runtime.h>

// ---------------------------------------------------------------------------
// 2-layer LSTM + FC head for gfx1250 (MI455X), wave32, WMMA bf16.
//
// Pipeline:
//   pack W_* -> bf16 WMMA B-fragments     (layout per CDNA5 ISA 7.12.2)
//   pack x   -> bf16 WMMA A-fragments
//   gemm_frag: XG0 = x @ Wih0^T + (bih0+bhh0)          [fp32, all timesteps]
//              A-fragments staged into LDS via TENSOR_LOAD_TO_LDS (TDM) when
//              the toolchain exposes the builtin; s_wait_tensorcnt sync.
//   lstm_scan: sequential over T, h kept packed in LDS, c in VGPRs,
//              per-step 16x2048 gate tile via v_wmma_f32_16x16x32_bf16
//   repack h-seq -> A-fragments, repeat for layer 1, then FC GEMM -> d_out
// ---------------------------------------------------------------------------

#define DEV __device__ __forceinline__

typedef __attribute__((ext_vector_type(16))) __bf16 v16bf;
typedef __attribute__((ext_vector_type(8)))  float  v8f;
typedef __attribute__((ext_vector_type(4)))  unsigned int u32x4;
typedef __attribute__((ext_vector_type(8)))  int    i32x8;
typedef __attribute__((ext_vector_type(4)))  int    i32x4;

// ---- Tensor Data Mover availability / arity detection ---------------------
#if defined(__has_builtin)
#  if __has_builtin(__builtin_amdgcn_tensor_load_to_lds) && \
      __has_builtin(__builtin_amdgcn_s_wait_tensorcnt)
#    define HAVE_TDM 1
#  endif
#endif
#ifndef HAVE_TDM
#  define HAVE_TDM 0
#endif
#if __has_include(<hip/amd_detail/amd_gfx1250_TDM.h>)
#  define TDM_6ARG 1   // amdgpu-toolchain (clang-23 / therock): 6-arg builtin
#else
#  define TDM_6ARG 0   // ROCm 7.2 (clang-22): 5-arg builtin
#endif

constexpr int Bsz = 64, T = 1024, IN = 256, H = 512, G4 = 2048, OUT = 128;
constexpr int Mtot   = Bsz * T;   // 65536 flattened (b,t) rows
constexpr int MT     = Mtot / 16; // 4096 M-tiles
constexpr int KT_IN  = IN / 32;   // 8  K-tiles (layer-0 input GEMM)
constexpr int KT_H   = H / 32;    // 16 K-tiles (H-dim reductions)
constexpr int NT_G   = G4 / 16;   // 128 N-tiles (gate dim)
constexpr int NT_OUT = OUT / 16;  // 8  N-tiles (FC head)

// ---- fp32 -> bf16 (round-to-nearest-even) --------------------------------
DEV unsigned short f2bf(float f) {
    unsigned int u = __float_as_uint(f);
    u += 0x7FFFu + ((u >> 16) & 1u);
    return (unsigned short)(u >> 16);
}

// ---- CDNA5 WMMA fragment maps (ISA 7.12.2, 16-bit operands) ---------------
// A (16x32, bf16): lane = half*16 + m ; element e -> k
DEV void a_map(int lane, int e, int& m, int& k) {
    int half = lane >> 4;
    m = lane & 15;
    int v = e >> 1, s = e & 1;
    k = (v < 4) ? (half * 8 + v * 2 + s) : (16 + half * 8 + (v - 4) * 2 + s);
}
// B (32x16, bf16): lane = khalf*16 + n ; element e -> k = e<8 ? khalf*8 + e : 16 + khalf*8 + (e-8)
DEV void b_map(int lane, int e, int& k, int& n) {
    n = lane & 15;
    k = (e < 8) ? ((lane >> 4) * 8 + e) : (16 + (lane >> 4) * 8 + (e - 8));
}
// C/D (16x16, f32): lane, reg r -> m = r + 8*(lane>>4), n = lane&15

DEV float sigm(float x) { return 1.0f / (1.0f + __expf(-x)); }
DEV float tanh_fast(float x) { return 2.0f / (1.0f + __expf(-2.0f * x)) - 1.0f; }

// ---------------------------------------------------------------------------
// Packing kernels
// ---------------------------------------------------------------------------
__global__ __launch_bounds__(256) void pack_a_f32(
    const float* __restrict__ src, unsigned short* __restrict__ dst,
    int KTt, int Ktot) {
    size_t tid = (size_t)blockIdx.x * blockDim.x + threadIdx.x;
    size_t total = (size_t)MT * KTt * 512;
    if (tid >= total) return;
    int e = tid & 15, lane = (tid >> 4) & 31;
    size_t fr = tid >> 9;
    int kt = (int)(fr % KTt);
    size_t mt = fr / KTt;
    int ml, kl; a_map(lane, e, ml, kl);
    size_t m = mt * 16 + ml; int k = kt * 32 + kl;
    dst[tid] = f2bf(src[m * (size_t)Ktot + k]);
}

__global__ __launch_bounds__(256) void pack_a_bf16(
    const unsigned short* __restrict__ src, unsigned short* __restrict__ dst,
    int KTt, int Ktot) {
    size_t tid = (size_t)blockIdx.x * blockDim.x + threadIdx.x;
    size_t total = (size_t)MT * KTt * 512;
    if (tid >= total) return;
    int e = tid & 15, lane = (tid >> 4) & 31;
    size_t fr = tid >> 9;
    int kt = (int)(fr % KTt);
    size_t mt = fr / KTt;
    int ml, kl; a_map(lane, e, ml, kl);
    size_t m = mt * 16 + ml; int k = kt * 32 + kl;
    dst[tid] = src[m * (size_t)Ktot + k];
}

// W stored [Ntot, Ktot] row-major (torch [out_features, in_features]); B = W^T
__global__ __launch_bounds__(256) void pack_b_f32(
    const float* __restrict__ W, unsigned short* __restrict__ dst,
    int KTt, int Ktot, int NTt) {
    size_t tid = (size_t)blockIdx.x * blockDim.x + threadIdx.x;
    size_t total = (size_t)NTt * KTt * 512;
    if (tid >= total) return;
    int e = tid & 15, lane = (tid >> 4) & 31;
    size_t fr = tid >> 9;
    int kt = (int)(fr % KTt);
    size_t nt = fr / KTt;
    int kl, nl; b_map(lane, e, kl, nl);
    size_t n = nt * 16 + nl; int k = kt * 32 + kl;
    dst[tid] = f2bf(W[n * (size_t)Ktot + k]);
}

__global__ __launch_bounds__(256) void vec_add(
    const float* __restrict__ a, const float* __restrict__ b,
    float* __restrict__ o, int n) {
    int i = blockIdx.x * blockDim.x + threadIdx.x;
    if (i < n) o[i] = a[i] + b[i];
}

// ---------------------------------------------------------------------------
// Fragment GEMM: C[Mtot x Ntot] (fp32, +bias) = Afrag x Bfrag
// grid = (MT, Ntot/128); one wave per block; wave covers 1 Mtile x 8 Ntiles.
// A-fragments for the whole K range of this M-tile are contiguous (KT x 1KB)
// -> stage them into LDS with one TDM descriptor per block when available.
// ---------------------------------------------------------------------------
__global__ __launch_bounds__(32) void gemm_frag(
    const unsigned short* __restrict__ Af, const unsigned short* __restrict__ Bf,
    const float* __restrict__ bias, float* __restrict__ C,
    int KT, int Ntot) {
    const int lane = threadIdx.x & 31;
    const int mt = blockIdx.x;
    const int ntg = blockIdx.y;  // group of 8 N-tiles
    const v16bf* Ap = (const v16bf*)Af;
    const v16bf* Bp = (const v16bf*)Bf;

#if HAVE_TDM
    // LDS staging buffer for up to KT_H = 16 A-fragments (16 KB).
    __shared__ __align__(128) unsigned short astage[KT_H * 512];
    {
        // D# group 0 (ISA 8.3): count=1, lds_addr=0 (astage is the only LDS
        // object), global_addr = &Af[mt*KT*512], type=2 ("image").
        unsigned long long ga =
            (unsigned long long)(uintptr_t)(Af + (size_t)mt * KT * 512);
        u32x4 g0 = { 1u,
                     0u,
                     (unsigned int)(ga & 0xFFFFFFFFu),
                     (unsigned int)((ga >> 32) & 0x01FFFFFFu) | (2u << 30) };
        // D# group 1 (ISA 8.4): data_size=8B, 1-D tile of nd0 = KT*128 units.
        unsigned int nd0 = (unsigned int)(KT * 128);
        i32x8 g1 = {
            (int)(3u << 16),                                  // data_size = 8B
            (int)((nd0 & 0xFFFFu) << 16),                     // tensor_dim0 lo
            (int)(((nd0 >> 16) & 0xFFFFu) | (1u << 16)),      // td0 hi | td1=1 lo
            (int)(nd0 << 16),                                 // td1 hi | tile_dim0
            1,                                                // tile_dim1=1, tile_dim2=0
            (int)nd0,                                         // tensor_dim0_stride lo
            0, 0 };
        i32x4 z4 = { 0, 0, 0, 0 };
#if TDM_6ARG
        i32x8 z8 = { 0, 0, 0, 0, 0, 0, 0, 0 };
        __builtin_amdgcn_tensor_load_to_lds(g0, g1, z4, z4, z8, 0);
#else
        __builtin_amdgcn_tensor_load_to_lds(g0, g1, z4, z4, 0);
#endif
        __builtin_amdgcn_s_wait_tensorcnt(0);
        __asm__ __volatile__("" ::: "memory");
    }
    const v16bf* ApL = (const v16bf*)astage;
#endif

    v8f acc[8];
#pragma unroll
    for (int j = 0; j < 8; ++j)
#pragma unroll
        for (int r = 0; r < 8; ++r) acc[j][r] = 0.0f;

    for (int kt = 0; kt < KT; ++kt) {
#if HAVE_TDM
        v16bf a = ApL[kt * 32 + lane];
#else
        v16bf a = Ap[(size_t)(mt * KT + kt) * 32 + lane];
        if (kt + 1 < KT)  // prefetch next-K A fragment (global_prefetch_b8)
            __builtin_prefetch((const void*)&Ap[(size_t)(mt * KT + kt + 1) * 32 + lane], 0, 0);
#endif
        if (kt + 1 < KT)  // prefetch next-K B fragment for first N-tile
            __builtin_prefetch(
                (const void*)&Bp[(size_t)((ntg * 8) * KT + kt + 1) * 32 + lane], 0, 0);
#pragma unroll
        for (int j = 0; j < 8; ++j) {
            int nt = ntg * 8 + j;
            v16bf b = Bp[(size_t)(nt * KT + kt) * 32 + lane];
            acc[j] = __builtin_amdgcn_wmma_f32_16x16x32_bf16(
                false, a, false, b, (short)0, acc[j], false, false);
        }
    }

    const int nlo = lane & 15, hi = lane >> 4;
#pragma unroll
    for (int j = 0; j < 8; ++j) {
        int n = (ntg * 8 + j) * 16 + nlo;
        float bv = bias ? bias[n] : 0.0f;
#pragma unroll
        for (int r = 0; r < 8; ++r) {
            int m = mt * 16 + r + hi * 8;
            C[(size_t)m * Ntot + n] = acc[j][r] + bv;
        }
    }
}

// ---------------------------------------------------------------------------
// LSTM recurrent scan. grid = B/16 blocks, 512 threads (16 waves).
// Block owns batch rows [bid*16, bid*16+16). h kept in LDS in packed
// A-fragment layout (16 K-tiles x 32 lanes x 16 bf16 = 16 KB). c in VGPRs.
// Wave w owns gate N-tiles {w + 16*j : j=0..7}; j pairs (i,f,g,o) share the
// same h-column per lane, so c/h updates are lane-local.
// ---------------------------------------------------------------------------
__global__ __launch_bounds__(512) void lstm_scan(
    const float* __restrict__ XG,            // [B, T, 4H] fp32 (bias included)
    const unsigned short* __restrict__ Whh,  // B-fragments [NT_G][KT_H][32][16]
    unsigned short* __restrict__ Hout) {     // [B, T, H] bf16 bits
    __shared__ __align__(64) unsigned short hp[KT_H * 32 * 16];  // 16 KB

    const int lane = threadIdx.x & 31;
    const int w = threadIdx.x >> 5;  // wave 0..15
    const int bbase = blockIdx.x * 16;
    const int nlo = lane & 15, hi = lane >> 4;
    const v16bf* Bp = (const v16bf*)Whh;

    for (int i = threadIdx.x; i < KT_H * 32 * 16; i += blockDim.x) hp[i] = 0;
    float c[2][8];
#pragma unroll
    for (int j = 0; j < 2; ++j)
#pragma unroll
        for (int r = 0; r < 8; ++r) c[j][r] = 0.0f;
    __syncthreads();

    for (int t = 0; t < T; ++t) {
        v8f acc[8];
        // gates init = precomputed input contribution
#pragma unroll
        for (int nidx = 0; nidx < 8; ++nidx) {
            int n = (w + nidx * 16) * 16 + nlo;
#pragma unroll
            for (int r = 0; r < 8; ++r) {
                int b = bbase + r + hi * 8;
                acc[nidx][r] = XG[((size_t)b * T + t) * G4 + n];
            }
        }
        // gates += h_{t-1} @ Whh^T   (16 K-tiles x 8 N-tiles WMMA per wave)
        for (int kt = 0; kt < KT_H; ++kt) {
            v16bf a = *(const v16bf*)&hp[(kt * 32 + lane) * 16];
#pragma unroll
            for (int nidx = 0; nidx < 8; ++nidx) {
                int nt = w + nidx * 16;
                v16bf b = Bp[(size_t)(nt * KT_H + kt) * 32 + lane];
                acc[nidx] = __builtin_amdgcn_wmma_f32_16x16x32_bf16(
                    false, a, false, b, (short)0, acc[nidx], false, false);
            }
        }
        __syncthreads();  // all waves done reading old h

        // activations; write new h packed into LDS + bf16 row to global
#pragma unroll
        for (int j = 0; j < 2; ++j) {
#pragma unroll
            for (int r = 0; r < 8; ++r) {
                float iv = sigm(acc[0 + j][r]);
                float fv = sigm(acc[2 + j][r]);
                float gv = tanh_fast(acc[4 + j][r]);
                float ov = sigm(acc[6 + j][r]);
                float cc = fv * c[j][r] + iv * gv;
                c[j][r] = cc;
                unsigned short hb = f2bf(ov * tanh_fast(cc));

                int hcol = (w + j * 16) * 16 + nlo;  // h column 0..511
                int m = r + hi * 8;                  // batch row 0..15
                // inverse A-map: (m, hcol%32) -> (lane', e') in tile hcol/32
                int kt2 = hcol >> 5, kl = hcol & 31;
                int half, v, s;
                if (kl < 16) { half = kl >> 3; int rem = kl & 7; v = rem >> 1; s = rem & 1; }
                else { int kk = kl - 16; half = kk >> 3; int rem = kk & 7; v = 4 + (rem >> 1); s = rem & 1; }
                hp[(kt2 * 32 + (half * 16 + m)) * 16 + (v * 2 + s)] = hb;
                Hout[((size_t)(bbase + m) * T + t) * H + hcol] = hb;
            }
        }
        __syncthreads();  // new h visible before next step
    }
}

// ---------------------------------------------------------------------------
extern "C" void kernel_launch(void* const* d_in, const int* in_sizes, int n_in,
                              void* d_out, int out_size, void* d_ws, size_t ws_size,
                              hipStream_t stream) {
    const float* x    = (const float*)d_in[0];
    const float* Wih0 = (const float*)d_in[1];
    const float* Whh0 = (const float*)d_in[2];
    const float* bih0 = (const float*)d_in[3];
    const float* bhh0 = (const float*)d_in[4];
    const float* Wih1 = (const float*)d_in[5];
    const float* Whh1 = (const float*)d_in[6];
    const float* bih1 = (const float*)d_in[7];
    const float* bhh1 = (const float*)d_in[8];
    const float* Wfc  = (const float*)d_in[9];
    const float* bfc  = (const float*)d_in[10];

    char* base = (char*)d_ws;
    size_t off = 0;
    auto take = [&](size_t bytes) -> char* {
        char* p = base + off;
        off = (off + bytes + 255) & ~(size_t)255;
        return p;
    };
    float*          XG     = (float*)take((size_t)Mtot * G4 * sizeof(float)); // 512 MB
    unsigned short* Afrag  = (unsigned short*)take((size_t)Mtot * H * 2);     // 64 MB
    unsigned short* Hplain = (unsigned short*)take((size_t)Mtot * H * 2);     // 64 MB
    unsigned short* WihF0  = (unsigned short*)take((size_t)G4 * IN * 2);
    unsigned short* WhhF0  = (unsigned short*)take((size_t)G4 * H * 2);
    unsigned short* WihF1  = (unsigned short*)take((size_t)G4 * H * 2);
    unsigned short* WhhF1  = (unsigned short*)take((size_t)G4 * H * 2);
    unsigned short* WfcF   = (unsigned short*)take((size_t)OUT * H * 2);
    float*          bsum0  = (float*)take(G4 * sizeof(float));
    float*          bsum1  = (float*)take(G4 * sizeof(float));

    auto blocks = [](size_t total, int bs) -> unsigned int {
        return (unsigned int)((total + bs - 1) / bs);
    };

    // weight / bias packing
    pack_b_f32<<<blocks((size_t)NT_G * KT_IN * 512, 256), 256, 0, stream>>>(Wih0, WihF0, KT_IN, IN, NT_G);
    pack_b_f32<<<blocks((size_t)NT_G * KT_H * 512, 256), 256, 0, stream>>>(Whh0, WhhF0, KT_H, H, NT_G);
    pack_b_f32<<<blocks((size_t)NT_G * KT_H * 512, 256), 256, 0, stream>>>(Wih1, WihF1, KT_H, H, NT_G);
    pack_b_f32<<<blocks((size_t)NT_G * KT_H * 512, 256), 256, 0, stream>>>(Whh1, WhhF1, KT_H, H, NT_G);
    pack_b_f32<<<blocks((size_t)NT_OUT * KT_H * 512, 256), 256, 0, stream>>>(Wfc, WfcF, KT_H, H, NT_OUT);
    vec_add<<<blocks(G4, 256), 256, 0, stream>>>(bih0, bhh0, bsum0, G4);
    vec_add<<<blocks(G4, 256), 256, 0, stream>>>(bih1, bhh1, bsum1, G4);

    // layer 0
    pack_a_f32<<<blocks((size_t)MT * KT_IN * 512, 256), 256, 0, stream>>>(x, Afrag, KT_IN, IN);
    gemm_frag<<<dim3(MT, NT_G / 8), 32, 0, stream>>>(Afrag, WihF0, bsum0, XG, KT_IN, G4);
    lstm_scan<<<Bsz / 16, 512, 0, stream>>>(XG, WhhF0, Hplain);

    // layer 1
    pack_a_bf16<<<blocks((size_t)MT * KT_H * 512, 256), 256, 0, stream>>>(Hplain, Afrag, KT_H, H);
    gemm_frag<<<dim3(MT, NT_G / 8), 32, 0, stream>>>(Afrag, WihF1, bsum1, XG, KT_H, G4);
    lstm_scan<<<Bsz / 16, 512, 0, stream>>>(XG, WhhF1, Hplain);

    // FC head -> d_out [B, T, OUT] fp32
    pack_a_bf16<<<blocks((size_t)MT * KT_H * 512, 256), 256, 0, stream>>>(Hplain, Afrag, KT_H, H);
    gemm_frag<<<dim3(MT, NT_OUT / 8), 32, 0, stream>>>(Afrag, WfcF, bfc, (float*)d_out, KT_H, OUT);
}
